// Attention_24008867185039
// MI455X (gfx1250) — matrix operand order repaired and checked by reading, not compiled, not run on hardware
//
#include <hip/hip_runtime.h>

// ---------------------------------------------------------------------------
// CDNA5 (gfx1250) fused multi-head self-attention, bf16 WMMA path + TDM.
//   kernel 0a: x (f32) -> xbf (bf16)
//   kernel 0b: W (f32, row-major [k][c]) -> Wt (bf16, transposed [c][k])
//   kernel 1 : QKV projection GEMM (TDM double-buffered LDS tiles)
//   kernel 2 : flash attention per (b,h) with v_wmma_f32_16x16x32_bf16
//   kernel 3 : output projection GEMM + bias (TDM double-buffered)
// ---------------------------------------------------------------------------

typedef __attribute__((ext_vector_type(16))) __bf16 v16bf;
typedef __attribute__((ext_vector_type(8)))  __bf16 v8bf;
typedef __attribute__((ext_vector_type(8)))  float  v8f;
typedef __attribute__((ext_vector_type(4)))  float  v4f;
typedef __attribute__((ext_vector_type(4)))  unsigned int v4u;
typedef __attribute__((ext_vector_type(8)))  int    v8i;
typedef __attribute__((ext_vector_type(4)))  int    v4i;

#define B_     2
#define N_     2048
#define DIM_   1024
#define HEADS_ 16
#define DHEAD_ 64
#define INNER_ 1024

#if defined(__has_builtin)
#if __has_builtin(__builtin_amdgcn_tensor_load_to_lds)
#define HAVE_TDM 1
#endif
#endif
#ifndef HAVE_TDM
#define HAVE_TDM 0
#endif

__device__ __forceinline__ v8f wmma_bf16(v16bf a, v16bf b, v8f c) {
  return __builtin_amdgcn_wmma_f32_16x16x32_bf16(false, a, false, b,
                                                 (short)0, c, false, false);
}

// A-matrix fragment (16x32 bf16): lane holds row m=lane&15, two contiguous
// 8-element runs at K = kOff + 8*hi and kOff + 16 + 8*hi (ISA 7.12.2).
__device__ __forceinline__ v16bf load_a_frag(const __bf16* rowBase,
                                             int rowStride, int kOff) {
  const int lane = threadIdx.x & 31;
  const __bf16* p = rowBase + (lane & 15) * rowStride + kOff + ((lane >> 4) << 3);
  v8bf lo = *(const v8bf*)(p);
  v8bf hi = *(const v8bf*)(p + 16);
  v16bf r;
#pragma unroll
  for (int i = 0; i < 8; ++i) { r[i] = lo[i]; r[8 + i] = hi[i]; }
  return r;
}

// B-matrix fragment (32x16 bf16): lane holds column n=lane&15, two contiguous
// 8-element runs at K = 8*hi and 16 + 8*hi.  colBase points at K-contiguous column storage.
__device__ __forceinline__ v16bf load_b_frag(const __bf16* colBase,
                                             int colStride) {
  const int lane = threadIdx.x & 31;
  const __bf16* p = colBase + (lane & 15) * colStride + ((lane >> 4) << 3);
  v8bf lo = *(const v8bf*)(p);
  v8bf hi = *(const v8bf*)(p + 16);
  v16bf r;
#pragma unroll
  for (int i = 0; i < 8; ++i) { r[i] = lo[i]; r[8 + i] = hi[i]; }
  return r;
}

#if HAVE_TDM
// Issue one TDM 2-D tile load global->LDS (bf16 elements), with LDS padding:
// tile rows of 2*tile_d0 bytes, pad 16B after each 2^(padIntCode+1) DWORDs.
// D# packing per CDNA5 ISA.  This toolchain's
// builtin takes 6 args: (g0 v4u, g1 v8i, g2 v4i, g3 v4i, v8i, i32 cpol).
__device__ __forceinline__ void tdm_load_2d(unsigned lds_off,
                                            const __bf16* gptr,
                                            unsigned tensor_d0,  // row len (elems)
                                            unsigned tensor_d1,  // #rows
                                            unsigned stride0,    // row stride (elems)
                                            unsigned tile_d0, unsigned tile_d1,
                                            unsigned padIntCode,
                                            unsigned padAmtCode) {
  unsigned long long ga = (unsigned long long)gptr;
  v4u g0;
  g0[0] = 1u;                                     // count=1 (valid), user mode
  g0[1] = lds_off;                                // lds_addr (bytes)
  g0[2] = (unsigned)(ga & 0xffffffffu);           // global_addr[31:0]
  g0[3] = (unsigned)((ga >> 32) & 0x01ffffffu)    // global_addr[56:32]
          | (2u << 30);                           // type=2 ("image")
  v8i g1;
  g1[0] = (int)((1u << 16)                        // data_size=1 -> 2 bytes
                | (1u << 20)                      // pad_enable
                | (padIntCode << 22)              // pad_interval
                | (padAmtCode << 25));            // pad_amount
  g1[1] = (int)((tensor_d0 & 0xffffu) << 16);     // tensor_dim0[15:0]
  g1[2] = (int)(((tensor_d0 >> 16) & 0xffffu) |
                ((tensor_d1 & 0xffffu) << 16));   // dim0 hi | dim1 lo
  g1[3] = (int)(((tensor_d1 >> 16) & 0xffffu) |
                ((tile_d0 & 0xffffu) << 16));     // dim1 hi | tile_dim0
  g1[4] = (int)(tile_d1 & 0xffffu);               // tile_dim1 (tile_dim2=0)
  g1[5] = (int)stride0;                           // tensor_dim0_stride[31:0]
  g1[6] = 0;                                      // stride hi, dim1_stride lo
  g1[7] = 0;
  v4i z4 = {0, 0, 0, 0};
  v8i z8 = {0, 0, 0, 0, 0, 0, 0, 0};
  __builtin_amdgcn_tensor_load_to_lds(g0, g1, z4, z4, z8, 0);
}
__device__ __forceinline__ unsigned lds_off_of(const void* p) {
  return (unsigned)(unsigned long long)p;         // low 32 bits = LDS byte addr
}
#endif

// ---------------------------------------------------------------------------
// Kernel 0a: f32 -> bf16 elementwise (x -> xbf).
// ---------------------------------------------------------------------------
__global__ __launch_bounds__(256) void cvt_bf16_kernel(
    const float* __restrict__ src, __bf16* __restrict__ dst, int n) {
  int i = (blockIdx.x * 256 + threadIdx.x) * 8;
  if (i + 7 < n) {
    v4f a = *(const v4f*)(src + i);
    v4f b = *(const v4f*)(src + i + 4);
    v8bf o;
#pragma unroll
    for (int j = 0; j < 4; ++j) { o[j] = (__bf16)a[j]; o[4 + j] = (__bf16)b[j]; }
    *(v8bf*)(dst + i) = o;
  }
}

// ---------------------------------------------------------------------------
// Kernel 0b: transpose + convert: dst[c][k] = (bf16)src[k][c].
// src is [1024][srcCols] f32; dst rows are relative to the passed base.
// ---------------------------------------------------------------------------
__global__ __launch_bounds__(256) void transpose_cvt_kernel(
    const float* __restrict__ src, int srcCols, __bf16* __restrict__ dst) {
  __shared__ __bf16 tile[32][33];
  const int tx = threadIdx.x & 31, ty = threadIdx.x >> 5;  // 32 x 8
  const int c0 = blockIdx.x * 32, k0 = blockIdx.y * 32;
#pragma unroll
  for (int i = 0; i < 32; i += 8)
    tile[ty + i][tx] = (__bf16)src[(size_t)(k0 + ty + i) * srcCols + c0 + tx];
  __syncthreads();
#pragma unroll
  for (int i = 0; i < 32; i += 8)
    dst[(size_t)(c0 + ty + i) * DIM_ + k0 + tx] = tile[tx][ty + i];
}

// ---------------------------------------------------------------------------
// Shared GEMM tile geometry: block tile 256(M) x 64(N), K-tile 64.
// 256 threads = 8 waves; wave computes 32 x 64 (2 row strips x 4 col tiles).
// LDS rows padded to 72 bf16 (144B): 16B-aligned, conflict-free for both
// fragment access patterns; matches TDM pad (32 DW data + 4 DW pad).
// ---------------------------------------------------------------------------
#define TST 72
#define PAD_INT_CODE 4u  /* 2^(4+1)=32 DWORDs = 128B of data ... */
#define PAD_AMT_CODE 3u  /* ... then 4 DWORDs = 16B of pad      */

// ---------------------------------------------------------------------------
// Kernel 1: QKV projection.  C[4096,3072] = xbf[4096,1024] @ Wt^T
// (Wt is [3072][1024] bf16, K-contiguous rows = perfect B-fragment feed).
// Epilogue routes each 64-col block (one head of Q/K/V); Q scaled by 2^-3.
// ---------------------------------------------------------------------------
__global__ __launch_bounds__(256) void qkv_proj_kernel(
    const __bf16* __restrict__ xbf, const __bf16* __restrict__ Wt,
    __bf16* __restrict__ Qbf, __bf16* __restrict__ Kbf,
    __bf16* __restrict__ Vt) {
  __shared__ __bf16 Abuf[2][256 * TST];
  __shared__ __bf16 Bbuf[2][64 * TST];
  const int tid = threadIdx.x, wave = tid >> 5, lane = tid & 31;
  const int hi = lane >> 4, n15 = lane & 15;
  const int row0 = blockIdx.x * 256;
  const int c0 = blockIdx.y * 64;

  v8f acc[2][4];
#pragma unroll
  for (int s = 0; s < 2; ++s)
#pragma unroll
    for (int j = 0; j < 4; ++j)
#pragma unroll
      for (int i = 0; i < 8; ++i) acc[s][j][i] = 0.f;

  const int NT = DIM_ / 64;
#if HAVE_TDM
  if (tid < 32) {
    tdm_load_2d(lds_off_of(&Abuf[0][0]), xbf + (size_t)row0 * DIM_,
                DIM_, 256, DIM_, 64, 256, PAD_INT_CODE, PAD_AMT_CODE);
    tdm_load_2d(lds_off_of(&Bbuf[0][0]), Wt + (size_t)c0 * DIM_,
                DIM_, 64, DIM_, 64, 64, PAD_INT_CODE, PAD_AMT_CODE);
  }
#endif
  for (int t = 0; t < NT; ++t) {
#if HAVE_TDM
    const int cur = t & 1;
    if (tid < 32) {
      if (t + 1 < NT) {
        const int nk = (t + 1) * 64, nb = (t + 1) & 1;
        tdm_load_2d(lds_off_of(&Abuf[nb][0]), xbf + (size_t)row0 * DIM_ + nk,
                    DIM_, 256, DIM_, 64, 256, PAD_INT_CODE, PAD_AMT_CODE);
        tdm_load_2d(lds_off_of(&Bbuf[nb][0]), Wt + (size_t)c0 * DIM_ + nk,
                    DIM_, 64, DIM_, 64, 64, PAD_INT_CODE, PAD_AMT_CODE);
        __builtin_amdgcn_s_wait_tensorcnt(2);   // previous tile complete
      } else {
        __builtin_amdgcn_s_wait_tensorcnt(0);   // last tile complete
      }
    }
    __syncthreads();
#else
    const int cur = 0;
    const int k0 = t * 64;
    // Fallback staging: plain bf16 vector copies (no conversion needed).
#pragma unroll
    for (int p = 0; p < 8; ++p) {             // A: 256x64 -> 2048 8-elem chunks
      int idx = p * 256 + tid;
      int r = idx >> 3, ch = (idx & 7) << 3;
      *(v8bf*)(&Abuf[0][r * TST + ch]) =
          *(const v8bf*)(xbf + (size_t)(row0 + r) * DIM_ + k0 + ch);
    }
#pragma unroll
    for (int p = 0; p < 2; ++p) {             // B: 64x64 -> 512 chunks
      int idx = p * 256 + tid;
      int r = idx >> 3, ch = (idx & 7) << 3;
      *(v8bf*)(&Bbuf[0][r * TST + ch]) =
          *(const v8bf*)(Wt + (size_t)(c0 + r) * DIM_ + k0 + ch);
    }
    __syncthreads();
#endif
    const __bf16* Ab = &Abuf[cur][0];
    const __bf16* Bb = &Bbuf[cur][0];
#pragma unroll
    for (int f = 0; f < 2; ++f) {
      v16bf a0 = load_a_frag(Ab + (wave * 32) * TST, TST, f * 32);
      v16bf a1 = load_a_frag(Ab + (wave * 32 + 16) * TST, TST, f * 32);
#pragma unroll
      for (int jt = 0; jt < 4; ++jt) {
        v16bf b = load_b_frag(Bb + (jt * 16) * TST + f * 32, TST);
        acc[0][jt] = wmma_bf16(a0, b, acc[0][jt]);
        acc[1][jt] = wmma_bf16(a1, b, acc[1][jt]);
      }
    }
    __syncthreads();
  }

  // Epilogue: route this 64-column block (== one head of Q, K, or V).
#pragma unroll
  for (int s = 0; s < 2; ++s) {
    const int gr = row0 + wave * 32 + s * 16;
#pragma unroll
    for (int jt = 0; jt < 4; ++jt) {
#pragma unroll
      for (int r = 0; r < 8; ++r) {
        int g = gr + r + 8 * hi;               // global row in [0, B*N)
        int b = g >> 11, n = g & (N_ - 1);
        int c = c0 + jt * 16 + n15;
        float v = acc[s][jt][r];
        if (c < INNER_) {
          int h = c >> 6, d = c & 63;
          Qbf[(((size_t)(b * HEADS_ + h)) * N_ + n) * DHEAD_ + d] =
              (__bf16)(v * 0.125f);            // fold 1/sqrt(64) (exact)
        } else if (c < 2 * INNER_) {
          int cc = c - INNER_; int h = cc >> 6, d = cc & 63;
          Kbf[(((size_t)(b * HEADS_ + h)) * N_ + n) * DHEAD_ + d] = (__bf16)v;
        } else {
          int cc = c - 2 * INNER_; int h = cc >> 6, d = cc & 63;
          Vt[(((size_t)(b * HEADS_ + h)) * DHEAD_ + d) * N_ + n] = (__bf16)v;
        }
      }
    }
  }
}

// ---------------------------------------------------------------------------
// Kernel 2: flash attention.  grid = (N/64, B*H); 128 threads = 4 waves;
// wave owns 16 query rows, iterates 64-wide key blocks with online softmax.
// ---------------------------------------------------------------------------
__global__ __launch_bounds__(128) void attn_kernel(
    const __bf16* __restrict__ Qbf, const __bf16* __restrict__ Kbf,
    const __bf16* __restrict__ Vt, __bf16* __restrict__ Obf) {
  constexpr int PST = 72;
  __shared__ __bf16 Pbuf[4][16 * PST];
  const int tid = threadIdx.x, wave = tid >> 5, lane = tid & 31;
  const int hi = lane >> 4, n15 = lane & 15;
  const int bh = blockIdx.y;
  const int b = bh >> 4, h = bh & (HEADS_ - 1);
  const int qbase = blockIdx.x * 64 + wave * 16;

  const __bf16* Qp = Qbf + ((size_t)bh * N_ + qbase) * DHEAD_;
  v16bf qf[2];
#pragma unroll
  for (int f = 0; f < 2; ++f) qf[f] = load_a_frag(Qp, DHEAD_, f * 32);

  float mrow[8], lrow[8];
  v8f o[4];
#pragma unroll
  for (int r = 0; r < 8; ++r) { mrow[r] = -1e30f; lrow[r] = 0.f; }
#pragma unroll
  for (int dt = 0; dt < 4; ++dt)
#pragma unroll
    for (int i = 0; i < 8; ++i) o[dt][i] = 0.f;

  __bf16* pw = &Pbuf[wave][0];

  for (int jb = 0; jb < N_; jb += 64) {
    v8f s[4];
#pragma unroll
    for (int jt = 0; jt < 4; ++jt)
#pragma unroll
      for (int i = 0; i < 8; ++i) s[jt][i] = 0.f;
#pragma unroll
    for (int f = 0; f < 2; ++f) {
#pragma unroll
      for (int jt = 0; jt < 4; ++jt) {
        const __bf16* kp =
            Kbf + ((size_t)bh * N_ + jb + jt * 16) * DHEAD_ + f * 32;
        v16bf bfr = load_b_frag(kp, DHEAD_);
        s[jt] = wmma_bf16(qf[f], bfr, s[jt]);
      }
    }
#pragma unroll
    for (int r = 0; r < 8; ++r) {
      float mx = s[0][r];
#pragma unroll
      for (int jt = 1; jt < 4; ++jt) mx = fmaxf(mx, s[jt][r]);
#pragma unroll
      for (int m = 1; m <= 8; m <<= 1) mx = fmaxf(mx, __shfl_xor(mx, m, 32));
      float mn = fmaxf(mrow[r], mx);
      float alpha = __expf(mrow[r] - mn);
      float sum = 0.f;
#pragma unroll
      for (int jt = 0; jt < 4; ++jt) {
        float p = __expf(s[jt][r] - mn);
        s[jt][r] = p;
        sum += p;
      }
#pragma unroll
      for (int m = 1; m <= 8; m <<= 1) sum += __shfl_xor(sum, m, 32);
      lrow[r] = lrow[r] * alpha + sum;
      mrow[r] = mn;
#pragma unroll
      for (int dt = 0; dt < 4; ++dt) o[dt][r] *= alpha;
    }
#pragma unroll
    for (int jt = 0; jt < 4; ++jt)
#pragma unroll
      for (int r = 0; r < 8; ++r)
        pw[(r + 8 * hi) * PST + jt * 16 + n15] = (__bf16)s[jt][r];

    v16bf pfr[2];
#pragma unroll
    for (int f = 0; f < 2; ++f) pfr[f] = load_a_frag(pw, PST, f * 32);
#pragma unroll
    for (int f = 0; f < 2; ++f) {
#pragma unroll
      for (int dt = 0; dt < 4; ++dt) {
        const __bf16* vp =
            Vt + ((size_t)bh * DHEAD_ + dt * 16) * N_ + jb + f * 32;
        v16bf bfr = load_b_frag(vp, N_);
        o[dt] = wmma_bf16(pfr[f], bfr, o[dt]);
      }
    }
  }
#pragma unroll
  for (int dt = 0; dt < 4; ++dt) {
#pragma unroll
    for (int r = 0; r < 8; ++r) {
      int q = qbase + r + 8 * hi;
      float val = o[dt][r] / lrow[r];
      Obf[((size_t)b * N_ + q) * INNER_ + h * DHEAD_ + dt * 16 + n15] =
          (__bf16)val;
    }
  }
}

// ---------------------------------------------------------------------------
// Kernel 3: output projection  out[4096,1024] = Obf @ Wot^T + bo  (fp32 out).
// Same TDM-double-buffered structure as kernel 1.
// ---------------------------------------------------------------------------
__global__ __launch_bounds__(256) void out_proj_kernel(
    const __bf16* __restrict__ Obf, const __bf16* __restrict__ Wot,
    const float* __restrict__ bo, float* __restrict__ out) {
  __shared__ __bf16 Abuf[2][256 * TST];
  __shared__ __bf16 Bbuf[2][64 * TST];
  const int tid = threadIdx.x, wave = tid >> 5, lane = tid & 31;
  const int hi = lane >> 4, n15 = lane & 15;
  const int row0 = blockIdx.x * 256;
  const int c0 = blockIdx.y * 64;

  v8f acc[2][4];
#pragma unroll
  for (int s = 0; s < 2; ++s)
#pragma unroll
    for (int j = 0; j < 4; ++j)
#pragma unroll
      for (int i = 0; i < 8; ++i) acc[s][j][i] = 0.f;

  const int NT = INNER_ / 64;
#if HAVE_TDM
  if (tid < 32) {
    tdm_load_2d(lds_off_of(&Abuf[0][0]), Obf + (size_t)row0 * INNER_,
                INNER_, 256, INNER_, 64, 256, PAD_INT_CODE, PAD_AMT_CODE);
    tdm_load_2d(lds_off_of(&Bbuf[0][0]), Wot + (size_t)c0 * DIM_,
                DIM_, 64, DIM_, 64, 64, PAD_INT_CODE, PAD_AMT_CODE);
  }
#endif
  for (int t = 0; t < NT; ++t) {
#if HAVE_TDM
    const int cur = t & 1;
    if (tid < 32) {
      if (t + 1 < NT) {
        const int nk = (t + 1) * 64, nb = (t + 1) & 1;
        tdm_load_2d(lds_off_of(&Abuf[nb][0]), Obf + (size_t)row0 * INNER_ + nk,
                    INNER_, 256, INNER_, 64, 256, PAD_INT_CODE, PAD_AMT_CODE);
        tdm_load_2d(lds_off_of(&Bbuf[nb][0]), Wot + (size_t)c0 * DIM_ + nk,
                    DIM_, 64, DIM_, 64, 64, PAD_INT_CODE, PAD_AMT_CODE);
        __builtin_amdgcn_s_wait_tensorcnt(2);   // previous tile complete
      } else {
        __builtin_amdgcn_s_wait_tensorcnt(0);   // last tile complete
      }
    }
    __syncthreads();
#else
    const int cur = 0;
    const int k0 = t * 64;
#pragma unroll
    for (int p = 0; p < 8; ++p) {
      int idx = p * 256 + tid;
      int r = idx >> 3, ch = (idx & 7) << 3;
      *(v8bf*)(&Abuf[0][r * TST + ch]) =
          *(const v8bf*)(Obf + (size_t)(row0 + r) * INNER_ + k0 + ch);
    }
#pragma unroll
    for (int p = 0; p < 2; ++p) {
      int idx = p * 256 + tid;
      int r = idx >> 3, ch = (idx & 7) << 3;
      *(v8bf*)(&Bbuf[0][r * TST + ch]) =
          *(const v8bf*)(Wot + (size_t)(c0 + r) * DIM_ + k0 + ch);
    }
    __syncthreads();
#endif
    const __bf16* Ab = &Abuf[cur][0];
    const __bf16* Bb = &Bbuf[cur][0];
#pragma unroll
    for (int f = 0; f < 2; ++f) {
      v16bf a0 = load_a_frag(Ab + (wave * 32) * TST, TST, f * 32);
      v16bf a1 = load_a_frag(Ab + (wave * 32 + 16) * TST, TST, f * 32);
#pragma unroll
      for (int jt = 0; jt < 4; ++jt) {
        v16bf b = load_b_frag(Bb + (jt * 16) * TST + f * 32, TST);
        acc[0][jt] = wmma_bf16(a0, b, acc[0][jt]);
        acc[1][jt] = wmma_bf16(a1, b, acc[1][jt]);
      }
    }
    __syncthreads();
  }

#pragma unroll
  for (int s = 0; s < 2; ++s) {
    const int gr = row0 + wave * 32 + s * 16;
#pragma unroll
    for (int jt = 0; jt < 4; ++jt) {
      float bias = bo[c0 + jt * 16 + n15];
#pragma unroll
      for (int r = 0; r < 8; ++r) {
        int g = gr + r + 8 * hi;
        out[(size_t)g * DIM_ + c0 + jt * 16 + n15] = acc[s][jt][r] + bias;
      }
    }
  }
}

// ---------------------------------------------------------------------------
// Host launcher.  Inputs: x, mask(all-true, ignored), Wq, Wkv, Wo, bo.
// Workspace: Qbf|Kbf|Vt|Obf (8MB each) | xbf (8MB) | Wt (6MB) | Wot (2MB).
// ---------------------------------------------------------------------------
extern "C" void kernel_launch(void* const* d_in, const int* in_sizes, int n_in,
                              void* d_out, int out_size, void* d_ws,
                              size_t ws_size, hipStream_t stream) {
  (void)in_sizes; (void)n_in; (void)out_size; (void)ws_size;
  const float* x   = (const float*)d_in[0];
  const float* Wq  = (const float*)d_in[2];
  const float* Wkv = (const float*)d_in[3];
  const float* Wo  = (const float*)d_in[4];
  const float* bo  = (const float*)d_in[5];
  float* out = (float*)d_out;

  char* ws = (char*)d_ws;
  const size_t SEG = (size_t)B_ * HEADS_ * N_ * DHEAD_ * sizeof(__bf16); // 8 MB
  __bf16* Qbf = (__bf16*)(ws);
  __bf16* Kbf = (__bf16*)(ws + SEG);
  __bf16* Vt  = (__bf16*)(ws + 2 * SEG);
  __bf16* Obf = (__bf16*)(ws + 3 * SEG);
  __bf16* xbf = (__bf16*)(ws + 4 * SEG);
  __bf16* Wt  = (__bf16*)(ws + 5 * SEG);                    // [3072][1024]
  __bf16* Wot = (__bf16*)(ws + 5 * SEG + (size_t)3072 * 1024 * 2);

  const int nx = B_ * N_ * DIM_;                            // 4M elements
  cvt_bf16_kernel<<<nx / (256 * 8), 256, 0, stream>>>(x, xbf, nx);
  transpose_cvt_kernel<<<dim3(32, 32), 256, 0, stream>>>(Wq, 1024, Wt);
  transpose_cvt_kernel<<<dim3(64, 32), 256, 0, stream>>>(
      Wkv, 2048, Wt + (size_t)1024 * 1024);
  transpose_cvt_kernel<<<dim3(32, 32), 256, 0, stream>>>(Wo, 1024, Wot);

  qkv_proj_kernel<<<dim3((B_ * N_) / 256, (3 * INNER_) / 64), 256, 0, stream>>>(
      xbf, Wt, Qbf, Kbf, Vt);
  attn_kernel<<<dim3(N_ / 64, B_ * HEADS_), 128, 0, stream>>>(
      Qbf, Kbf, Vt, Obf);
  out_proj_kernel<<<dim3((B_ * N_) / 256, DIM_ / 64), 256, 0, stream>>>(
      Obf, Wot, bo, out);
}
